// Graph_module_net_0_loss_type_18631568130084
// MI455X (gfx1250) — matrix operand order repaired and checked by reading, not compiled, not run on hardware
//
#include <hip/hip_runtime.h>
#include <hip/hip_bf16.h>
#include <stdint.h>

#define B_      16
#define N_      1024
#define CIN_    256
#define MID_    512
#define OUT_    256
#define G_      4
#define CHILDS_ 512

typedef __attribute__((ext_vector_type(16))) __bf16 v16bf;
typedef __attribute__((ext_vector_type(8)))  float  v8f;

union U16x16 { uint4 q[2]; unsigned short u[16]; v16bf v; };

__device__ __forceinline__ unsigned short f2bf(float f) {
  unsigned int x = __float_as_uint(f);
  x += 0x7fffu + ((x >> 16) & 1u);          // round-to-nearest-even
  return (unsigned short)(x >> 16);
}

// ---------------- WMMA operand loaders (contiguous k order, same in A and B) ----------------
// A-matrix 16x32: lane L -> row m = L&15; lanes 0-15 hold K 0..15,
// lanes 16-31 hold K 16..31  => one contiguous 16-element run.
__device__ __forceinline__ v16bf load_a_bf16(const unsigned short* A, int lda, int m0, int k0, int lane) {
  int m = m0 + (lane & 15);
  int half = (lane >> 4) & 1;
  const unsigned short* p = A + (size_t)m * lda + k0 + half * 16;
  U16x16 t;
  t.q[0] = *(const uint4*)(p);
  t.q[1] = *(const uint4*)(p + 8);
  return t.v;
}
// B-matrix 32x16: lane L -> col n = L&15; lanes 0-15 K=0..15, lanes 16-31 K=16..31.
// B source layout: Brow[n][k] (row per output column) => one contiguous 16-run.
__device__ __forceinline__ v16bf load_b_bf16(const unsigned short* Brow, int ldb, int n0, int k0, int lane) {
  int n = n0 + (lane & 15);
  int half = (lane >> 4) & 1;
  const unsigned short* p = Brow + (size_t)n * ldb + k0 + half * 16;
  U16x16 t;
  t.q[0] = *(const uint4*)(p);
  t.q[1] = *(const uint4*)(p + 8);
  return t.v;
}
// B operand out of an LDS-staged 32x32 bf16 tile (row stride 32 elements).
__device__ __forceinline__ v16bf load_b_lds(const unsigned short* sB, int ntile, int lane) {
  int n = ntile * 16 + (lane & 15);
  int half = (lane >> 4) & 1;
  const unsigned short* p = sB + n * 32 + half * 16;
  U16x16 t;
  t.q[0] = *(const uint4*)(p);
  t.q[1] = *(const uint4*)(p + 8);
  return t.v;
}

__device__ __forceinline__ v8f wmma_bf16(v16bf a, v16bf b, v8f c) {
  return __builtin_amdgcn_wmma_f32_16x16x32_bf16(false, a, false, b, (short)0, c, false, false);
}

// C-tile: lane L holds col n = L&15, rows m0 + (L>=16 ? 8 : 0) + r, r=0..7.
__device__ __forceinline__ void store_c_rowmajor_f32(float* C, int ldc, int m0, int n0, int lane, v8f c) {
  int n = n0 + (lane & 15);
  int mb = m0 + ((lane >> 4) & 1) * 8;
#pragma unroll
  for (int r = 0; r < 8; ++r) C[(size_t)(mb + r) * ldc + n] = c[r];
}
__device__ __forceinline__ void store_c_rowmajor_bf16(unsigned short* C, int ldc, int m0, int n0, int lane, v8f c) {
  int n = n0 + (lane & 15);
  int mb = m0 + ((lane >> 4) & 1) * 8;
#pragma unroll
  for (int r = 0; r < 8; ++r) C[(size_t)(mb + r) * ldc + n] = f2bf(c[r]);
}
// Transposed store: contiguous 8 floats per lane at T[n][mb..mb+7].
__device__ __forceinline__ void store_c_transposed_f32(float* T, int ldt, int m0, int n0, int lane, v8f c) {
  int n = n0 + (lane & 15);
  int mb = m0 + ((lane >> 4) & 1) * 8;
  float* p = T + (size_t)n * ldt + mb;
  *(float4*)(p)     = make_float4(c[0], c[1], c[2], c[3]);
  *(float4*)(p + 4) = make_float4(c[4], c[5], c[6], c[7]);
}

__device__ __forceinline__ float block_sum(float v, float* sred, int tid) {
  sred[tid] = v; __syncthreads();
  for (int off = 128; off > 0; off >>= 1) {
    if (tid < off) sred[tid] += sred[tid + off];
    __syncthreads();
  }
  float r = sred[0]; __syncthreads();
  return r;
}

// ---------------------------------- kernels ----------------------------------
__global__ void k_init(unsigned int* col) {
  int j = blockIdx.x * blockDim.x + threadIdx.x;
  if (j < N_) col[j] = 0u;
}

// Vectorized f32 -> bf16 conversion (8 elements/thread).
__global__ void __launch_bounds__(256) k_cvt(const float* s, unsigned short* d, int count) {
  int i = (blockIdx.x * blockDim.x + threadIdx.x) * 8;
  if (i >= count) return;
  float4 f0 = *(const float4*)(s + i);
  float4 f1 = *(const float4*)(s + i + 4);
  union { unsigned short u[8]; uint4 q; } o;
  o.u[0] = f2bf(f0.x); o.u[1] = f2bf(f0.y); o.u[2] = f2bf(f0.z); o.u[3] = f2bf(f0.w);
  o.u[4] = f2bf(f1.x); o.u[5] = f2bf(f1.y); o.u[6] = f2bf(f1.z); o.u[7] = f2bf(f1.w);
  *(uint4*)(d + i) = o.q;
}

__global__ void k_weights(const float* Wg, const float* W1, const float* W2,
                          unsigned short* WgT, unsigned short* W1b, unsigned short* W2b) {
  int t = blockIdx.x * blockDim.x + threadIdx.x;
  if (t < CIN_ * OUT_) {                 // WgT[o][c] = Wg[c][o]
    int o = t / CIN_, c = t % CIN_;
    WgT[t] = f2bf(Wg[c * OUT_ + o]);
  }
  if (t < MID_ * (CIN_ / G_)) W1b[t] = f2bf(W1[t]);
  if (t < OUT_ * (MID_ / G_)) W2b[t] = f2bf(W2[t]);
}

__global__ void __launch_bounds__(256) k_ljli(const float* x, const float* W_att, float* lj, float* li) {
  int lane = threadIdx.x & 31;
  int wave = threadIdx.x >> 5;
  int row = blockIdx.x * 8 + wave;            // b*N + n
  const float* xr = x + (size_t)row * CIN_;
  float aq = 0.f, ak = 0.f;
  for (int c = lane; c < CIN_; c += 32) {
    float xv = xr[c];
    aq += xv * W_att[c];                      // Wq (H==1)
    ak += xv * W_att[CIN_ + c];               // Wk
  }
  for (int off = 16; off > 0; off >>= 1) {
    aq += __shfl_xor(aq, off, 32);
    ak += __shfl_xor(ak, off, 32);
  }
  if (lane == 0) { lj[row] = aq; li[row] = ak; }
}

__global__ void __launch_bounds__(256) k_topk(const float* m1g, const float* m2g, const float* score,
                                              const float* lj, const float* li, const float* b_att,
                                              unsigned int* col) {
  __shared__ float sAv[N_];
  __shared__ float sM1[N_];
  __shared__ float sM2[N_];
  __shared__ int   sIdx[N_];
  __shared__ float sVal[N_];
  __shared__ int   sScan[256];
  __shared__ int   sBase;
  int tid = threadIdx.x;
  int b = blockIdx.x / N_;
  float liv = li[blockIdx.x];
  float ba  = b_att[0];
  const float* m1r = m1g + (size_t)blockIdx.x * N_;
  const float* m2r = m2g + (size_t)blockIdx.x * N_;
  const float* smr = score + (size_t)b * N_;
  const float* ljr = lj + (size_t)b * N_;
  for (int j = tid; j < N_; j += 256) {
    float sm = smr[j];
    float a  = 1.f / (1.f + __expf(-(liv + ljr[j] + ba)));
    sAv[j] = a;
    sM1[j] = m1r[j] * sm;
    sM2[j] = m2r[j] * sm;
  }
  __syncthreads();
  for (int pass = 0; pass < 2; ++pass) {
    const float* flg = (pass == 0) ? sM1 : sM2;
    int ktop = (pass == 0) ? 128 : 256;
    const int kbot = 128;
    if (tid == 0) sBase = 0;
    __syncthreads();
    // ordered compaction of positives (preserves ascending index order)
    for (int chunk = 0; chunk < N_ / 256; ++chunk) {
      int j = chunk * 256 + tid;
      float v = sAv[j] * flg[j];
      int p = (v != 0.f) ? 1 : 0;
      sScan[tid] = p;
      __syncthreads();
      for (int off = 1; off < 256; off <<= 1) {
        int t = (tid >= off) ? sScan[tid - off] : 0;
        __syncthreads();
        sScan[tid] += t;
        __syncthreads();
      }
      int incl = sScan[tid];
      int base = sBase;
      if (p) { sIdx[base + incl - 1] = j; sVal[base + incl - 1] = v; }
      __syncthreads();
      if (tid == 255) sBase = base + incl;
      __syncthreads();
    }
    int P = sBase;
    // positives: exact rank among positives (zeros rank below all positives)
    for (int p = tid; p < P; p += 256) {
      float v = sVal[p]; int id = sIdx[p];
      int r = 0;
      for (int q = 0; q < P; ++q) {
        float vq = sVal[q];
        r += (vq > v) || (vq == v && sIdx[q] < id);
      }
      if (r < ktop) col[id] = 1u;
    }
    // zeros: bottom-k picks first kbot zeros; top-k fills leftover slots with
    // lowest-index zeros (lax.top_k ties break toward lower index)
    for (int j = tid; j < N_; j += 256) {
      if (sAv[j] * flg[j] == 0.f) {
        int lo = 0, hi = P;                    // count positives with idx < j
        while (lo < hi) { int mid = (lo + hi) >> 1; if (sIdx[mid] < j) lo = mid + 1; else hi = mid; }
        int zr = j - lo;
        if (zr < kbot || P + zr < ktop) col[j] = 1u;
      }
    }
    __syncthreads();
  }
}

__global__ void __launch_bounds__(256) k_amat(const float* m1g, const float* m2g, const float* score,
                                              const float* lj, const float* li, const float* b_att,
                                              const unsigned int* col, unsigned short* Amat) {
  int tid = threadIdx.x;
  int row = blockIdx.x;                      // b*N + i
  int b = row / N_;
  int i = row % N_;
  float liv = li[row];
  float ba  = b_att[0];
  float fdiag = (score[(size_t)b * N_ + i] == 0.f) ? 1.f : 0.f;
  const float* m1r = m1g + (size_t)row * N_;
  const float* m2r = m2g + (size_t)row * N_;
  const float* smr = score + (size_t)b * N_;
  const float* ljr = lj + (size_t)b * N_;
  unsigned short* arow = Amat + (size_t)row * N_;
  const float inv = 1.f / (float)CHILDS_;
  for (int j = tid; j < N_; j += 256) {
    float sm = smr[j];
    float a  = 1.f / (1.f + __expf(-(liv + ljr[j] + ba)));
    float m  = sm * (m1r[j] + m2r[j]);
    float cv = col[j] ? 1.f : 0.f;
    float val = a * m * cv;
    if (j == i) val += fdiag;
    arow[j] = f2bf(val * inv);
  }
}

// gts: (16384 x 256) = gt_bf16 (16384 x 256) @ WgT^T, relu(+bg). 1x4 tiles/wave.
__global__ void __launch_bounds__(256) k_gts(const unsigned short* gtb, const unsigned short* WgT,
                                             const float* bg, float* outg) {
  int lane = threadIdx.x & 31;
  int wave = threadIdx.x >> 5;
  int tile = blockIdx.x * 8 + wave;          // 1024 m-tiles x 4 n-groups
  int m0 = (tile >> 2) * 16;
  int n0 = (tile & 3) * 64;
  v8f acc[4] = {};
  for (int k0 = 0; k0 < CIN_; k0 += 32) {
    v16bf a = load_a_bf16(gtb, CIN_, m0, k0, lane);
#pragma unroll
    for (int t = 0; t < 4; ++t) {
      v16bf bb = load_b_bf16(WgT, CIN_, n0 + t * 16, k0, lane);
      acc[t] = wmma_bf16(a, bb, acc[t]);
    }
  }
#pragma unroll
  for (int t = 0; t < 4; ++t) {
    float bias = bg[n0 + t * 16 + (lane & 15)];
#pragma unroll
    for (int r = 0; r < 8; ++r) { float v = acc[t][r] + bias; acc[t][r] = v > 0.f ? v : 0.f; }
    store_c_rowmajor_f32(outg, OUT_, m0, n0 + t * 16, lane, acc[t]);
  }
}

// grouped conv1: o1[b][o][n] = relu(W1[o][:] . x[b][n][g*64..]) ; 1x4 tiles/wave.
__global__ void __launch_bounds__(256) k_gconv1(const unsigned short* xb, const unsigned short* W1b, const float* b1,
                                                unsigned short* o1b, float* o1T) {
  int lane = threadIdx.x & 31;
  int wave = threadIdx.x >> 5;
  int tile = blockIdx.x * 8 + wave;          // b*512 + ot*16 + ng
  int b = tile >> 9;
  int rem = tile & 511;
  int m0 = (rem >> 4) * 16;                  // o in [0,512)
  int n0 = (rem & 15) * 64;
  int g = m0 >> 7;                           // 128 outputs per group
  const unsigned short* xbb = xb + (size_t)b * N_ * CIN_;
  v8f acc[4] = {};
  for (int k0 = 0; k0 < CIN_ / G_; k0 += 32) {
    v16bf a = load_a_bf16(W1b, CIN_ / G_, m0, k0, lane);
#pragma unroll
    for (int t = 0; t < 4; ++t) {
      v16bf bb = load_b_bf16(xbb, CIN_, n0 + t * 16, g * (CIN_ / G_) + k0, lane);
      acc[t] = wmma_bf16(a, bb, acc[t]);
    }
  }
  int mb = m0 + ((lane >> 4) & 1) * 8;
#pragma unroll
  for (int t = 0; t < 4; ++t) {
#pragma unroll
    for (int r = 0; r < 8; ++r) { float v = acc[t][r] + b1[mb + r]; acc[t][r] = v > 0.f ? v : 0.f; }
    store_c_rowmajor_bf16(o1b + (size_t)b * MID_ * N_, N_, m0, n0 + t * 16, lane, acc[t]);
    store_c_transposed_f32(o1T + (size_t)b * N_ * MID_, MID_, m0, n0 + t * 16, lane, acc[t]);
  }
}

// Unified big matmul:  CT[b][n][m] = sum_k A[b][m][k] * Bm[b][n][k]  (bf16 x bf16 -> f32)
// Block = 8 waves: 128 m-rows x 32 n-cols. The 32x32 bf16 B tile is staged to LDS
// by wave 0 with CDNA5 async-to-LDS copies (ASYNCcnt), double-buffered.
__global__ void __launch_bounds__(256) k_mm(const unsigned short* A, const unsigned short* Bm, float* CT,
                                            int ldct, int mblocks, int nblocks,
                                            size_t aStride, size_t ctStride) {
  __shared__ unsigned short sB[2][32 * 32];   // single shared array -> LDS offset 0
  int lane = threadIdx.x & 31;
  int wave = threadIdx.x >> 5;
  int perb = mblocks * nblocks;
  int b   = blockIdx.x / perb;
  int rem = blockIdx.x % perb;
  int mb  = rem / nblocks;
  int nb  = rem % nblocks;
  int m0  = mb * 128 + wave * 16;
  int n0  = nb * 32;
  const unsigned short* Ab = A  + (size_t)b * aStride;
  const unsigned short* Bb = Bm + (size_t)b * N_ * N_ + (size_t)n0 * N_;
  float* Cb = CT + (size_t)b * ctStride;

  // stage chunk c (32 k's of 32 rows) into buffer buf: 4 async b128 ops, wave 0 only
  auto stage = [&](int c, int buf) {
#pragma unroll
    for (int t = 0; t < 4; ++t) {
      int r   = t * 8 + (lane >> 2);
      int seg = lane & 3;
      const unsigned short* ga = Bb + (size_t)r * N_ + c * 32 + seg * 8;
      unsigned la = (unsigned)(buf * 2048 + r * 64 + seg * 16);
      asm volatile("global_load_async_to_lds_b128 %0, %1, off"
                   :: "v"(la), "v"(ga) : "memory");
    }
  };

  v8f acc0 = {}, acc1 = {};
  if (wave == 0) stage(0, 0);
  const int C = N_ / 32;
  for (int c = 0; c < C; ++c) {
    int buf = c & 1;
    if (wave == 0) asm volatile("s_wait_asynccnt 0x0" ::: "memory");
    __syncthreads();                         // chunk c visible; prev buffer free
    if (wave == 0 && c + 1 < C) stage(c + 1, buf ^ 1);
    v16bf a  = load_a_bf16(Ab, N_, m0, c * 32, lane);
    v16bf b0 = load_b_lds(&sB[buf][0], 0, lane);
    v16bf b1 = load_b_lds(&sB[buf][0], 1, lane);
    acc0 = wmma_bf16(a, b0, acc0);
    acc1 = wmma_bf16(a, b1, acc1);
  }
  store_c_transposed_f32(Cb, ldct, m0, n0,      lane, acc0);
  store_c_transposed_f32(Cb, ldct, m0, n0 + 16, lane, acc1);
}

__global__ void __launch_bounds__(256) k_ln1(const float* o1mT, const float* o1T, const float* g1, const float* beta1,
                                             unsigned short* o1pT) {
  __shared__ float sred[256];
  int tid = threadIdx.x;
  size_t base = (size_t)blockIdx.x * MID_;
  float x0 = o1mT[base + tid];
  float x1 = o1mT[base + 256 + tid];
  float mean = block_sum(x0 + x1, sred, tid) * (1.f / MID_);
  float d0 = x0 - mean, d1 = x1 - mean;
  float var = block_sum(d0 * d0 + d1 * d1, sred, tid) * (1.f / MID_);
  float rstd = rsqrtf(var + 1e-6f);
  float y0 = d0 * rstd * g1[tid] + beta1[tid];
  float y1 = d1 * rstd * g1[256 + tid] + beta1[256 + tid];
  o1pT[base + tid]       = f2bf(o1T[base + tid] + y0);
  o1pT[base + 256 + tid] = f2bf(o1T[base + 256 + tid] + y1);
}

// grouped conv2 from transposed bf16 activations; 1x4 tiles/wave.
__global__ void __launch_bounds__(256) k_gconv2(const unsigned short* o1pT, const unsigned short* W2b, const float* b2,
                                                unsigned short* o2b, float* o2T) {
  int lane = threadIdx.x & 31;
  int wave = threadIdx.x >> 5;
  int tile = blockIdx.x * 8 + wave;          // b*256 + ot*16 + ng
  int b = tile >> 8;
  int rem = tile & 255;
  int m0 = (rem >> 4) * 16;                  // o in [0,256)
  int n0 = (rem & 15) * 64;
  int g = m0 >> 6;                           // 64 outputs per group
  const unsigned short* Bb = o1pT + (size_t)b * N_ * MID_;
  v8f acc[4] = {};
  for (int k0 = 0; k0 < MID_ / G_; k0 += 32) {
    v16bf a = load_a_bf16(W2b, MID_ / G_, m0, k0, lane);
#pragma unroll
    for (int t = 0; t < 4; ++t) {
      v16bf bb = load_b_bf16(Bb, MID_, n0 + t * 16, g * (MID_ / G_) + k0, lane);
      acc[t] = wmma_bf16(a, bb, acc[t]);
    }
  }
  int mb = m0 + ((lane >> 4) & 1) * 8;
#pragma unroll
  for (int t = 0; t < 4; ++t) {
#pragma unroll
    for (int r = 0; r < 8; ++r) { float v = acc[t][r] + b2[mb + r]; acc[t][r] = v > 0.f ? v : 0.f; }
    store_c_rowmajor_bf16(o2b + (size_t)b * OUT_ * N_, N_, m0, n0 + t * 16, lane, acc[t]);
    store_c_transposed_f32(o2T + (size_t)b * N_ * OUT_, OUT_, m0, n0 + t * 16, lane, acc[t]);
  }
}

__global__ void __launch_bounds__(256) k_final(const float* o2mT, const float* o2T, const float* g2, const float* beta2,
                                               float* out2, float* outnode) {
  __shared__ float sred[256];
  int tid = threadIdx.x;
  size_t base = (size_t)blockIdx.x * OUT_;
  float x0 = o2mT[base + tid];
  float mean = block_sum(x0, sred, tid) * (1.f / OUT_);
  float d0 = x0 - mean;
  float var = block_sum(d0 * d0, sred, tid) * (1.f / OUT_);
  float rstd = rsqrtf(var + 1e-6f);
  float nf = d0 * rstd * g2[tid] + beta2[tid];
  outnode[base + tid] = nf;
  out2[base + tid] = o2T[base + tid] + nf;
}

// --------------------------------- launcher ---------------------------------
extern "C" void kernel_launch(void* const* d_in, const int* in_sizes, int n_in,
                              void* d_out, int out_size, void* d_ws, size_t ws_size,
                              hipStream_t stream) {
  const float* x     = (const float*)d_in[0];
  const float* m1    = (const float*)d_in[1];
  const float* m2    = (const float*)d_in[2];
  const float* score = (const float*)d_in[3];
  const float* gt    = (const float*)d_in[4];
  const float* W_att = (const float*)d_in[5];
  const float* b_att = (const float*)d_in[6];
  const float* W1    = (const float*)d_in[7];
  const float* b1    = (const float*)d_in[8];
  const float* W2    = (const float*)d_in[9];
  const float* b2    = (const float*)d_in[10];
  const float* g1    = (const float*)d_in[11];
  const float* beta1 = (const float*)d_in[12];
  const float* g2    = (const float*)d_in[13];
  const float* beta2 = (const float*)d_in[14];
  const float* Wg    = (const float*)d_in[15];
  const float* bg    = (const float*)d_in[16];

  float* out = (float*)d_out;
  const size_t S = (size_t)B_ * N_ * OUT_;
  float* out2    = out;            // output2   (B,N,OUT)
  float* outg    = out + S;        // gts       (B,N,OUT)
  float* outnode = out + 2 * S;    // node_feat (B,N,OUT)

  char* ws = (char*)d_ws;
  size_t off = 0;
  auto alloc = [&](size_t bytes) -> char* {
    char* p = ws + off;
    off = (off + bytes + 255) & ~(size_t)255;
    return p;
  };
  float*          lj   = (float*)alloc((size_t)B_ * N_ * 4);
  float*          li   = (float*)alloc((size_t)B_ * N_ * 4);
  unsigned int*   col  = (unsigned int*)alloc((size_t)N_ * 4);
  unsigned short* WgT  = (unsigned short*)alloc((size_t)OUT_ * CIN_ * 2);
  unsigned short* W1b  = (unsigned short*)alloc((size_t)MID_ * (CIN_ / G_) * 2);
  unsigned short* W2b  = (unsigned short*)alloc((size_t)OUT_ * (MID_ / G_) * 2);
  unsigned short* xb   = (unsigned short*)alloc((size_t)B_ * N_ * CIN_ * 2);
  unsigned short* gtb  = (unsigned short*)alloc((size_t)B_ * N_ * CIN_ * 2);
  unsigned short* Amat = (unsigned short*)alloc((size_t)B_ * N_ * N_ * 2);
  unsigned short* o1b  = (unsigned short*)alloc((size_t)B_ * MID_ * N_ * 2);
  float*          o1T  = (float*)alloc((size_t)B_ * N_ * MID_ * 4);
  float*          o1mT = (float*)alloc((size_t)B_ * N_ * MID_ * 4);
  unsigned short* o1pT = (unsigned short*)alloc((size_t)B_ * N_ * MID_ * 2);
  unsigned short* o2b  = (unsigned short*)alloc((size_t)B_ * OUT_ * N_ * 2);
  float*          o2T  = (float*)alloc((size_t)B_ * N_ * OUT_ * 4);
  float*          o2mT = (float*)alloc((size_t)B_ * N_ * OUT_ * 4);

  const int ACT = B_ * N_ * CIN_;            // 4,194,304

  k_init   <<<(N_ + 255) / 256, 256, 0, stream>>>(col);
  k_weights<<<(CIN_ * OUT_ + 255) / 256, 256, 0, stream>>>(Wg, W1, W2, WgT, W1b, W2b);
  k_cvt    <<<ACT / (256 * 8), 256, 0, stream>>>(x,  xb,  ACT);
  k_cvt    <<<ACT / (256 * 8), 256, 0, stream>>>(gt, gtb, ACT);
  k_ljli   <<<B_ * N_ / 8, 256, 0, stream>>>(x, W_att, lj, li);
  k_topk   <<<B_ * N_, 256, 0, stream>>>(m1, m2, score, lj, li, b_att, col);
  k_amat   <<<B_ * N_, 256, 0, stream>>>(m1, m2, score, lj, li, b_att, col, Amat);
  k_gts    <<<(B_ * N_ / 16) * 4 / 8, 256, 0, stream>>>(gtb, WgT, bg, outg);
  k_gconv1 <<<B_ * 512 / 8, 256, 0, stream>>>(xb, W1b, b1, o1b, o1T);
  // o1m: M=512 -> 4 m-blocks of 128; N=1024 -> 32 n-blocks of 32
  k_mm     <<<B_ * 4 * 32, 256, 0, stream>>>(o1b, Amat, o1mT, MID_, 4, 32,
                                             (size_t)MID_ * N_, (size_t)N_ * MID_);
  k_ln1    <<<B_ * N_, 256, 0, stream>>>(o1mT, o1T, g1, beta1, o1pT);
  k_gconv2 <<<B_ * 256 / 8, 256, 0, stream>>>(o1pT, W2b, b2, o2b, o2T);
  // o2m: M=256 -> 2 m-blocks of 128
  k_mm     <<<B_ * 2 * 32, 256, 0, stream>>>(o2b, Amat, o2mT, OUT_, 2, 32,
                                             (size_t)OUT_ * N_, (size_t)N_ * OUT_);
  k_final  <<<B_ * N_, 256, 0, stream>>>(o2mT, o2T, g2, beta2, out2, outnode);
}
